// LSTMModel_71227737637223
// MI455X (gfx1250) — matrix operand order repaired and checked by reading, not compiled, not run on hardware
//
#include <hip/hip_runtime.h>
#include <hip/hip_bf16.h>

typedef __attribute__((ext_vector_type(16))) __bf16       v16bf;
typedef __attribute__((ext_vector_type(8)))  float        v8f;
typedef __attribute__((ext_vector_type(8)))  unsigned int v8u;
typedef __attribute__((ext_vector_type(4)))  unsigned int v4u;

constexpr int BATCH = 512;
constexpr int HD    = 1024;
constexpr int VOC   = 50257;

// f32 pair -> packed bf16 (lo in [15:0], hi in [31:16]); round-half-up,
// fused with a single v_perm_b32 (no 16-bit subregister moves).
__device__ __forceinline__ unsigned pack_bf16(float lo, float hi) {
    unsigned a = __float_as_uint(lo) + 0x8000u;
    unsigned b = __float_as_uint(hi) + 0x8000u;
    // result bytes {3,2,1,0} = {b.3, b.2, a.3, a.2}
    return __builtin_amdgcn_perm(b, a, 0x07060302u);
}

// ---------------------------------------------------------------------------
// C[M,N] = sum_p A_p[M,K] @ B_p[K,N]  (+ bias0 + bias1), row-major.
// A ld = K, B ld = ldb, C ld = N. M multiple of 64, K multiple of 32.
// Block tile 64x256, 8 waves (2x4), wave tile 32x64 -> 8 WMMA / K-step.
// Double-buffered LDS + register prefetch: one barrier per K-step.
// LDS tiles are fragment-major so every WMMA fragment = 2x ds_load_b128.
// ---------------------------------------------------------------------------
__global__ __launch_bounds__(256) void gemm_bf16_wmma(
    const float* __restrict__ A0, const float* __restrict__ B0,
    const float* __restrict__ A1, const float* __restrict__ B1,
    const float* __restrict__ bias0, const float* __restrict__ bias1,
    float* __restrict__ C, int N, int K, int ldb)
{
    constexpr int BK = 32;
    constexpr int PA = 20;                    // dword pitch (16 used), 80B: 16B-aligned quads
    constexpr int PB = 20;
    __shared__ unsigned ldsA[2][64 * PA];     // [buf][m][kpair]
    __shared__ unsigned ldsB[2][256 * PB];    // [buf][n][kpair] (fragment-major)

    const int tid    = threadIdx.x;
    const int lane   = tid & 31;
    const int wave   = tid >> 5;
    const int wm     = wave >> 2;             // 0..1
    const int wn     = wave & 3;              // 0..3
    const int lane15 = lane & 15;
    const int hi16   = lane >> 4;

    const long m0 = (long)blockIdx.x * 64;
    const long n0 = (long)blockIdx.y * 256;

    const int  colB = (int)n0 + tid;
    const bool cok  = (colB < N);

    const int stepsPerPass = K / BK;
    const int nsteps = ((A1 != nullptr) ? 2 : 1) * stepsPerPass;

    // ---- register-resident prefetch of one 64x32 A tile + 32x256 B tile
    float2 aReg[4];
    float  bLo[16], bHi[16];

    auto fetch = [&](int st) {
        const bool second = (st >= stepsPerPass);        // no integer division
        const int  k0     = (second ? st - stepsPerPass : st) * BK;
        const float* __restrict__ A  = second ? A1 : A0;
        const float* __restrict__ Bm = second ? B1 : B0;
        #pragma unroll
        for (int r = 0; r < 4; ++r) {
            int e = tid + r * 256;            // 0..1023
            int m = e >> 4, kp = e & 15;
            aReg[r] = *(const float2*)(A + (m0 + m) * K + k0 + 2 * kp);
        }
        #pragma unroll
        for (int kp = 0; kp < 16; ++kp) {
            float lo = 0.f, hi = 0.f;
            if (cok) {
                lo = Bm[(long)(k0 + 2 * kp)     * ldb + colB];
                hi = Bm[(long)(k0 + 2 * kp + 1) * ldb + colB];
            }
            bLo[kp] = lo; bHi[kp] = hi;
        }
        if (cok && (k0 + BK < K))             // gfx1250 global_prefetch of next K-panel
            __builtin_prefetch(Bm + (long)(k0 + BK) * ldb + colB, 0, 1);
    };

    auto stage = [&](int buf) {
        #pragma unroll
        for (int r = 0; r < 4; ++r) {
            int e = tid + r * 256;
            int m = e >> 4, kp = e & 15;
            ldsA[buf][m * PA + kp] = pack_bf16(aReg[r].x, aReg[r].y);
        }
        #pragma unroll
        for (int q = 0; q < 4; ++q) {
            v4u pk;
            #pragma unroll
            for (int j = 0; j < 4; ++j)
                pk[j] = pack_bf16(bLo[q * 4 + j], bHi[q * 4 + j]);
            *(v4u*)(&ldsB[buf][tid * PB + q * 4]) = pk;
        }
    };

    fetch(0);
    stage(0);
    __syncthreads();

    v8f acc[2][4] = {};

    for (int st = 0; st < nsteps; ++st) {
        const int cur = st & 1;
        if (st + 1 < nsteps) fetch(st + 1);   // loads in flight under the WMMAs

        // ---- A fragments: contiguous lane dwords -> 2x ds_load_b128 each
        v16bf af[2];
        #pragma unroll
        for (int ms = 0; ms < 2; ++ms) {
            const unsigned* ap =
                &ldsA[cur][(wm * 32 + ms * 16 + lane15) * PA + hi16 * 4];
            v4u lo = *(const v4u*)ap;
            v4u hi = *(const v4u*)(ap + 8);
            af[ms] = __builtin_bit_cast(
                v16bf, __builtin_shufflevector(lo, hi, 0,1,2,3,4,5,6,7));
        }
        // ---- 4 B fragments x 2 M-subtiles = 8 WMMA
        #pragma unroll
        for (int s = 0; s < 4; ++s) {
            const unsigned* bp =
                &ldsB[cur][(wn * 64 + s * 16 + lane15) * PB + hi16 * 4];
            v4u lo = *(const v4u*)bp;
            v4u hi = *(const v4u*)(bp + 8);
            v16bf bf = __builtin_bit_cast(
                v16bf, __builtin_shufflevector(lo, hi, 0,1,2,3,4,5,6,7));
            #pragma unroll
            for (int ms = 0; ms < 2; ++ms)
                acc[ms][s] = __builtin_amdgcn_wmma_f32_16x16x32_bf16(
                    false, af[ms], false, bf, (short)0, acc[ms][s],
                    false, false);
        }

        if (st + 1 < nsteps) {
            stage(cur ^ 1);                   // write other buffer; readers use cur
            __syncthreads();                  // single barrier per K-step
        }
    }

    // ---- epilogue: C/D layout -> lanes 0-15: M=v, 16-31: M=8+v
    #pragma unroll
    for (int ms = 0; ms < 2; ++ms) {
        const long mrow = m0 + wm * 32 + ms * 16 + hi16 * 8;
        #pragma unroll
        for (int s = 0; s < 4; ++s) {
            int col = (int)n0 + wn * 64 + s * 16 + lane15;
            if (col < N) {
                float bias = 0.f;
                if (bias0) bias += bias0[col];
                if (bias1) bias += bias1[col];
                #pragma unroll
                for (int v = 0; v < 8; ++v)
                    C[(mrow + v) * (long)N + col] = acc[ms][s][v] + bias;
            }
        }
    }
}

// ---------------------------------------------------------------------------
__global__ __launch_bounds__(256) void gather_emb(
    const int* __restrict__ x, const float* __restrict__ emb,
    float* __restrict__ out)
{
    const int b = blockIdx.x;
    const long row = (long)x[b];
    const float4* src = (const float4*)(emb + row * HD);
    float4*       dst = (float4*)(out + (long)b * HD);
    dst[threadIdx.x] = src[threadIdx.x];
}

// gates layout per batch row: [i | f | o | g] each HD wide
__global__ __launch_bounds__(256) void lstm_pointwise(
    const float* __restrict__ gates, const float* __restrict__ cprev,
    float* __restrict__ ncout, float* __restrict__ nhout)
{
    const int idx = blockIdx.x * 256 + threadIdx.x;   // < BATCH*HD
    const int b = idx >> 10;
    const int n = idx & (HD - 1);
    const float* g = gates + (long)b * (4 * HD);
    float i = g[n], f = g[n + HD], o = g[n + 2 * HD], gg = g[n + 3 * HD];
    float is = 1.f / (1.f + __expf(-i));
    float fs = 1.f / (1.f + __expf(-f));
    float os = 1.f / (1.f + __expf(-o));
    float gt = tanhf(gg);
    float nc = fs * cprev[idx] + is * gt;
    float nh = os * tanhf(nc);
    ncout[idx] = nc;
    nhout[idx] = nh;
}

__global__ __launch_bounds__(256) void logsoftmax_rows(float* __restrict__ out)
{
    __shared__ float red[256];
    const int tid = threadIdx.x;
    float* p = out + (long)blockIdx.x * VOC;

    float m = -3.4e38f;
    for (int i = tid; i < VOC; i += 256) m = fmaxf(m, p[i]);
    red[tid] = m; __syncthreads();
    for (int s = 128; s > 0; s >>= 1) {
        if (tid < s) red[tid] = fmaxf(red[tid], red[tid + s]);
        __syncthreads();
    }
    m = red[0]; __syncthreads();

    float sum = 0.f;
    for (int i = tid; i < VOC; i += 256) sum += __expf(p[i] - m);
    red[tid] = sum; __syncthreads();
    for (int s = 128; s > 0; s >>= 1) {
        if (tid < s) red[tid] += red[tid + s];
        __syncthreads();
    }
    const float lse = m + __logf(red[0]);
    __syncthreads();

    for (int i = tid; i < VOC; i += 256) p[i] -= lse;
}

// ---------------------------------------------------------------------------
extern "C" void kernel_launch(void* const* d_in, const int* in_sizes, int n_in,
                              void* d_out, int out_size, void* d_ws, size_t ws_size,
                              hipStream_t stream)
{
    (void)in_sizes; (void)n_in; (void)out_size; (void)ws_size;
    const int*   x   = (const int*)  d_in[0];
    const float* c0  = (const float*)d_in[1];
    const float* h0  = (const float*)d_in[2];
    const float* c1  = (const float*)d_in[3];
    const float* h1  = (const float*)d_in[4];
    const float* emb = (const float*)d_in[5];
    const float* W   = (const float*)d_in[6];
    const float* Wb  = (const float*)d_in[7];
    const float* U   = (const float*)d_in[8];
    const float* Ub  = (const float*)d_in[9];
    const float* fcW = (const float*)d_in[10];
    const float* fcb = (const float*)d_in[11];

    float* logp = (float*)d_out;                       // [512, 50257]
    float* nc0  = logp + (long)BATCH * VOC;            // [512,1024] each
    float* nh0  = nc0 + BATCH * HD;
    float* nc1  = nh0 + BATCH * HD;
    float* nh1  = nc1 + BATCH * HD;

    float* inp0  = (float*)d_ws;                       // [512,1024]
    float* gates = inp0 + BATCH * HD;                  // [512,4096]

    const long wstride = (long)HD * 4 * HD;            // per-layer weight stride

    gather_emb<<<BATCH, 256, 0, stream>>>(x, emb, inp0);

    // layer 0: gates = inp0@W0 + h0@U0 + Wb0 + Ub0
    gemm_bf16_wmma<<<dim3(BATCH / 64, (4 * HD) / 256), 256, 0, stream>>>(
        inp0, W, h0, U, Wb, Ub, gates, 4 * HD, HD, 4 * HD);
    lstm_pointwise<<<(BATCH * HD) / 256, 256, 0, stream>>>(gates, c0, nc0, nh0);

    // layer 1: gates = nh0@W1 + h1@U1 + Wb1 + Ub1
    gemm_bf16_wmma<<<dim3(BATCH / 64, (4 * HD) / 256), 256, 0, stream>>>(
        nh0, W + wstride, h1, U + wstride, Wb + 4 * HD, Ub + 4 * HD,
        gates, 4 * HD, HD, 4 * HD);
    lstm_pointwise<<<(BATCH * HD) / 256, 256, 0, stream>>>(gates, c1, nc1, nh1);

    // fc: logits = nh1@fcW + fcb  (N = 50257, guarded)
    gemm_bf16_wmma<<<dim3(BATCH / 64, (VOC + 255) / 256), 256, 0, stream>>>(
        nh1, fcW, nullptr, nullptr, fcb, nullptr, logp, VOC, HD, VOC);

    logsoftmax_rows<<<BATCH, 256, 0, stream>>>(logp);
}
